// RgbNet_90117003805324
// MI455X (gfx1250) — matrix operand order repaired and checked by reading, not compiled, not run on hardware
//
#include <hip/hip_runtime.h>

typedef __attribute__((ext_vector_type(16))) _Float16 v16h;
typedef __attribute__((ext_vector_type(8)))  _Float16 v8h;
typedef __attribute__((ext_vector_type(8)))  float    v8f;

#define N_RAYS     16384
#define N_SAMPLES  300
#define NUM_LEVELS 6
#define LEVEL_DIM  4
#define HASH_SZ    (1u << 19)
#define HASH_MASK  (HASH_SZ - 1u)

// ---------------------------------------------------------------------------
// Kernel 0: convert + pad weights to f16, contiguous block:
//   [W1p 64x128 | W2p 64x64 | W3p 64x64 | W4p 16x64]  = 17408 halfs
// ---------------------------------------------------------------------------
__global__ __launch_bounds__(256) void cvt_kernel(
    const float* __restrict__ W1, const float* __restrict__ W2,
    const float* __restrict__ W3, const float* __restrict__ W4,
    _Float16* __restrict__ wbase) {
  int t = blockIdx.x * 256 + threadIdx.x;
  if (t < 64 * 128) {
    int n = t >> 7, k = t & 127;
    wbase[t] = (_Float16)(k < 120 ? W1[n * 120 + k] : 0.0f);
  } else if (t < 8192 + 4096) {
    int u = t - 8192;
    wbase[t] = (_Float16)W2[u];
  } else if (t < 8192 + 4096 + 4096) {
    int u = t - 12288;
    wbase[t] = (_Float16)W3[u];
  } else if (t < 8192 + 4096 + 4096 + 1024) {
    int u = t - 16384;
    int n = u >> 6, k = u & 63;
    wbase[t] = (_Float16)(n < 3 ? W4[n * 64 + k] : 0.0f);
  }
}

// ---------------------------------------------------------------------------
// Kernel 1: hash-grid features at the 5 gathered samples only.
// One thread per (ray, offset d in 0..4).  feat layout: [ray][128] f16,
// feat[ray][d*24 + l*4 + c], k=120..127 zero-padded.
// ---------------------------------------------------------------------------
__global__ __launch_bounds__(256) void feat_kernel(
    const float* __restrict__ x, const int* __restrict__ idxf,
    const float* __restrict__ emb, _Float16* __restrict__ feat) {
  int tid = blockIdx.x * 256 + threadIdx.x;
  if (tid >= N_RAYS * 5) return;
  int ray = tid / 5;
  int d   = tid - ray * 5;          // 0..4 -> delta d-2

  int s = idxf[ray] + (d - 2);
  s = s < 0 ? 0 : (s > N_SAMPLES - 1 ? N_SAMPLES - 1 : s);
  float t = (float)s * (1.0f / (float)(N_SAMPLES - 1));

  float4 xr = ((const float4*)x)[ray];
  float p[3];
  p[0] = xr.x * (1.0f - t) + xr.z * t;
  p[1] = xr.y * (1.0f - t) + xr.w * t;
  p[2] = t;

  float fout[NUM_LEVELS * LEVEL_DIM];

#pragma unroll
  for (int l = 0; l < NUM_LEVELS; ++l) {
    const int R = 4 << l;
    float w[3];
    int   p0[3];
#pragma unroll
    for (int a = 0; a < 3; ++a) {
      float pos = p[a] * (float)R;
      float fl  = floorf(pos);
      p0[a] = (int)fl;
      w[a]  = pos - fl;
    }
    float acc0 = 0.f, acc1 = 0.f, acc2 = 0.f, acc3 = 0.f;
#pragma unroll
    for (int cn = 0; cn < 8; ++cn) {
      // _OFFS order: dim0 = bit2, dim1 = bit1, dim2 = bit0
      int o0 = (cn >> 2) & 1, o1 = (cn >> 1) & 1, o2 = cn & 1;
      int c0 = p0[0] + o0; c0 = c0 < 0 ? 0 : (c0 > R ? R : c0);
      int c1 = p0[1] + o1; c1 = c1 < 0 ? 0 : (c1 > R ? R : c1);
      int c2 = p0[2] + o2; c2 = c2 < 0 ? 0 : (c2 > R ? R : c2);
      float cw = (o0 ? w[0] : 1.0f - w[0]) *
                 (o1 ? w[1] : 1.0f - w[1]) *
                 (o2 ? w[2] : 1.0f - w[2]);
      unsigned int idx;
      if (l < 5) {
        idx = (unsigned int)(c0 + c1 * (R + 1) + c2 * (R + 1) * (R + 1));
      } else {
        idx = (((unsigned int)c0 * 1u) ^
               ((unsigned int)c1 * 2654435761u) ^
               ((unsigned int)c2 * 805459861u)) & HASH_MASK;
      }
      const float4 e = *(const float4*)(emb + ((size_t)l * HASH_SZ + idx) * 4);
      acc0 += cw * e.x; acc1 += cw * e.y; acc2 += cw * e.z; acc3 += cw * e.w;
    }
    fout[l * 4 + 0] = acc0; fout[l * 4 + 1] = acc1;
    fout[l * 4 + 2] = acc2; fout[l * 4 + 3] = acc3;
  }

  _Float16* fp = feat + (size_t)ray * 128 + d * 24;
#pragma unroll
  for (int i = 0; i < 24; ++i) fp[i] = (_Float16)fout[i];
  if (d == 0) {
    _Float16* zp = feat + (size_t)ray * 128 + 120;
#pragma unroll
    for (int i = 0; i < 8; ++i) zp[i] = (_Float16)0.0f;
  }
}

// ---------------------------------------------------------------------------
// WMMA fragment helpers (16-bit, contiguous k order shared by A and B, wave32).
//  A (16xK tile, ld-major): lane = m + 16*hf; elems i -> K = kb+16*hf+i.
//  B (Kx16 tile):           lane = n + 16*hf; elems i -> K = kb+16*hf+i.
// ---------------------------------------------------------------------------
__device__ __forceinline__ v16h load_a(const _Float16* base, int ld, int kb,
                                       int m, int hf) {
  const _Float16* rp = base + (size_t)m * ld + kb + hf * 16;
  v8h lo = *(const v8h*)rp;
  v8h hi = *(const v8h*)(rp + 8);
  return __builtin_shufflevector(lo, hi, 0, 1, 2, 3, 4, 5, 6, 7,
                                         8, 9, 10, 11, 12, 13, 14, 15);
}

__device__ __forceinline__ v16h load_b(const _Float16* base, int ld, int kb,
                                       int n, int hf) {
  return *(const v16h*)(base + (size_t)n * ld + kb + hf * 16);
}

__device__ __forceinline__ v8f wmma16(v16h a, v16h b, v8f c) {
  return __builtin_amdgcn_wmma_f32_16x16x32_f16(false, a, false, b,
                                                (short)0, c, false, false);
}

// ---------------------------------------------------------------------------
// Kernel 2: fused MLP.  128 threads = 4 wave32; each wave owns a 16-ray tile.
// Weights (f16, 34KB) bulk-copied to LDS once per block; activations ping-
// pong through per-wave 16x64 f16 LDS tiles with fused bias+ReLU epilogues.
// ---------------------------------------------------------------------------
__global__ __launch_bounds__(128) void mlp_kernel(
    const _Float16* __restrict__ feat, const _Float16* __restrict__ wbase,
    const float* __restrict__ b1, const float* __restrict__ b2,
    const float* __restrict__ b3, const float* __restrict__ b4,
    float* __restrict__ out) {
  __shared__ __align__(32) _Float16 wsm[17408];          // 34 KB weights
  __shared__ __align__(32) _Float16 hbuf[2][4][16 * 64]; // 16 KB activations

  const int wv   = threadIdx.x >> 5;
  const int lane = threadIdx.x & 31;
  const int hf   = lane >> 4;
  const int rs   = lane & 15;

  // cooperative weight copy (contiguous, dword granularity)
  {
    const unsigned int* src = (const unsigned int*)wbase;
    unsigned int*       dst = (unsigned int*)wsm;
    for (int i = threadIdx.x; i < 17408 / 2; i += 128) dst[i] = src[i];
  }
  __syncthreads();

  const _Float16* sW1 = wsm;          // 64 x 128
  const _Float16* sW2 = wsm + 8192;   // 64 x 64
  const _Float16* sW3 = wsm + 12288;  // 64 x 64
  const _Float16* sW4 = wsm + 16384;  // 16 x 64

  const int tile = blockIdx.x * 4 + wv;   // 1024 tiles exactly
  const int row0 = tile * 16;
  const _Float16* A0 = feat + (size_t)row0 * 128;

  _Float16* h0 = &hbuf[0][wv][0];
  _Float16* h1 = &hbuf[1][wv][0];

  // ---- layer 1: [16x128] x [128x64] -----------------------------------
  v8f c1[4] = {v8f{}, v8f{}, v8f{}, v8f{}};
#pragma unroll
  for (int kt = 0; kt < 4; ++kt) {
    v16h a = load_a(A0, 128, kt * 32, rs, hf);
#pragma unroll
    for (int nt = 0; nt < 4; ++nt)
      c1[nt] = wmma16(a, load_b(sW1, 128, kt * 32, nt * 16 + rs, hf), c1[nt]);
  }
#pragma unroll
  for (int nt = 0; nt < 4; ++nt) {
    int n = nt * 16 + rs;
    float bv = b1[n];
#pragma unroll
    for (int r = 0; r < 8; ++r) {
      int m = r + 8 * hf;
      h0[m * 64 + n] = (_Float16)fmaxf(c1[nt][r] + bv, 0.0f);
    }
  }
  __syncthreads();

  // ---- layer 2: [16x64] x [64x64] --------------------------------------
  v8f c2[4] = {v8f{}, v8f{}, v8f{}, v8f{}};
#pragma unroll
  for (int kt = 0; kt < 2; ++kt) {
    v16h a = load_a(h0, 64, kt * 32, rs, hf);
#pragma unroll
    for (int nt = 0; nt < 4; ++nt)
      c2[nt] = wmma16(a, load_b(sW2, 64, kt * 32, nt * 16 + rs, hf), c2[nt]);
  }
#pragma unroll
  for (int nt = 0; nt < 4; ++nt) {
    int n = nt * 16 + rs;
    float bv = b2[n];
#pragma unroll
    for (int r = 0; r < 8; ++r) {
      int m = r + 8 * hf;
      h1[m * 64 + n] = (_Float16)fmaxf(c2[nt][r] + bv, 0.0f);
    }
  }
  __syncthreads();

  // ---- layer 3: [16x64] x [64x64] --------------------------------------
  v8f c3[4] = {v8f{}, v8f{}, v8f{}, v8f{}};
#pragma unroll
  for (int kt = 0; kt < 2; ++kt) {
    v16h a = load_a(h1, 64, kt * 32, rs, hf);
#pragma unroll
    for (int nt = 0; nt < 4; ++nt)
      c3[nt] = wmma16(a, load_b(sW3, 64, kt * 32, nt * 16 + rs, hf), c3[nt]);
  }
#pragma unroll
  for (int nt = 0; nt < 4; ++nt) {
    int n = nt * 16 + rs;
    float bv = b3[n];
#pragma unroll
    for (int r = 0; r < 8; ++r) {
      int m = r + 8 * hf;
      h0[m * 64 + n] = (_Float16)fmaxf(c3[nt][r] + bv, 0.0f);
    }
  }
  __syncthreads();

  // ---- layer 4: [16x64] x [64x16pad] -> 3 valid cols -------------------
  v8f c4 = v8f{};
#pragma unroll
  for (int kt = 0; kt < 2; ++kt) {
    v16h a = load_a(h0, 64, kt * 32, rs, hf);
    c4 = wmma16(a, load_b(sW4, 64, kt * 32, rs, hf), c4);
  }
  if (rs < 3) {
    float bv = b4[rs];
#pragma unroll
    for (int r = 0; r < 8; ++r) {
      int m = r + 8 * hf;
      out[(size_t)(row0 + m) * 3 + rs] = c4[r] + bv;
    }
  }
}

// ---------------------------------------------------------------------------
extern "C" void kernel_launch(void* const* d_in, const int* in_sizes, int n_in,
                              void* d_out, int out_size, void* d_ws,
                              size_t ws_size, hipStream_t stream) {
  const float* x    = (const float*)d_in[0];
  const int*   idxf = (const int*)d_in[1];
  const float* emb  = (const float*)d_in[2];
  const float* W1   = (const float*)d_in[3];
  const float* b1   = (const float*)d_in[4];
  const float* W2   = (const float*)d_in[5];
  const float* b2   = (const float*)d_in[6];
  const float* W3   = (const float*)d_in[7];
  const float* b3   = (const float*)d_in[8];
  const float* W4   = (const float*)d_in[9];
  const float* b4   = (const float*)d_in[10];
  float* out = (float*)d_out;

  // workspace: feat f16 [16384][128] then contiguous f16 weights
  _Float16* featp = (_Float16*)d_ws;
  _Float16* wbase = featp + (size_t)N_RAYS * 128;

  cvt_kernel<<<68, 256, 0, stream>>>(W1, W2, W3, W4, wbase);
  feat_kernel<<<(N_RAYS * 5) / 256, 256, 0, stream>>>(x, idxf, emb, featp);
  mlp_kernel<<<N_RAYS / 64, 128, 0, stream>>>(featp, wbase, b1, b2, b3, b4, out);
}
